// LSTM_82703890252379
// MI455X (gfx1250) — matrix operand order repaired and checked by reading, not compiled, not run on hardware
//
#include <hip/hip_runtime.h>

typedef _Float16 f16;
typedef __attribute__((ext_vector_type(16))) _Float16 v16h;
typedef __attribute__((ext_vector_type(8)))  float    v8f;
typedef __attribute__((ext_vector_type(4)))  int      v4i;

#define HID 32
#define TSTEPS 1024
#define BATCH 2048
#define TILE_B 16        // batch rows per block (WMMA M)
#define NGATE_TILES 4    // gate tiles per wave after the 2-way split

union AB32 { v16h h; v4i q[2]; };

__device__ __forceinline__ float tanh_fast(float x) {
#if __has_builtin(__builtin_amdgcn_tanhf)
  return __builtin_amdgcn_tanhf(x);
#else
  float y;
  asm("v_tanh_f32 %0, %1" : "=v"(y) : "v"(x));
  return y;
#endif
}
__device__ __forceinline__ float sigmoid_fast(float x) {
  return __builtin_fmaf(0.5f, tanh_fast(0.5f * x), 0.5f);
}

// A fragment (16x32 f16) from LDS: two contiguous 16-byte chunks
__device__ __forceinline__ v16h lds_load_afrag(const f16* p) {
  AB32 u;
  u.q[0] = *(const v4i*)(p);
  u.q[1] = *(const v4i*)(p + 8);   // +16 bytes
  return u.h;
}

// B fragment (32x16 f16) built once from global f32 weights:
// lane carries row g of W ([128,32] row-major), K = kbase..kbase+15
__device__ __forceinline__ v16h gload_bfrag(const float* __restrict__ W,
                                            int g, int kbase) {
  const float* row = W + g * HID + kbase;
  v16h r;
#pragma unroll
  for (int e = 0; e < 16; ++e) r[e] = (f16)row[e];
  return r;
}

__device__ __forceinline__ v8f wmma_f16(v16h a, v16h b, v8f c) {
  // D = A(16x32 f16) * B(32x16 f16) + C(16x16 f32)
  return __builtin_amdgcn_wmma_f32_16x16x32_f16(false, a, false, b, (short)0, c,
                                                false, false);
}

__global__ __launch_bounds__(64, 1)
void lstm2_fused_wmma(const float* __restrict__ x,
                      const float* __restrict__ W_ih0,
                      const float* __restrict__ W_hh0,
                      const float* __restrict__ b_ih0,
                      const float* __restrict__ b_hh0,
                      const float* __restrict__ W_ih1,
                      const float* __restrict__ W_hh1,
                      const float* __restrict__ b_ih1,
                      const float* __restrict__ b_hh1,
                      const float* __restrict__ W_fc,
                      const float* __restrict__ b_fc,
                      float* __restrict__ out)
{
  // per-block staging: h transpose buffers (shared by the 2 cooperating waves)
  __shared__ __align__(64) f16  sH0[TILE_B * HID];   // 1 KB
  __shared__ __align__(64) f16  sH1[TILE_B * HID];   // 1 KB
  __shared__ float sRed[2][TILE_B];                  // FC cross-wave partials

  const int tid  = threadIdx.x;
  const int w    = tid >> 5;             // wave 0: hidden 0..15, wave 1: hidden 16..31
  const int lane = tid & 31;
  const int gl   = lane & 15;            // column within 16-wide tiles
  const int hb   = (lane < 16) ? 0 : 8;  // row base for C/D vgpr mapping
  const int kb   = (lane < 16) ? 0 : 16; // K base for B fragments

  const int b0 = blockIdx.x * TILE_B;    // both waves share this batch tile
  const float* xrow = x + (size_t)(b0 + gl) * TSTEPS;

  // ---- gate tiles owned by this wave: j=0:i, 1:f, 2:g, 3:o  (n = w + 2j) ----
  // weight B-fragments live in registers for the whole T-loop (12 x v16h)
  v16h whh0F[NGATE_TILES], whh1F[NGATE_TILES], wih1F[NGATE_TILES];
  float wx[NGATE_TILES], bb0[NGATE_TILES];
  float b1h[NGATE_TILES];                // layer-1 bias, pre-scaled for tanh fold
#pragma unroll
  for (int j = 0; j < NGATE_TILES; ++j) {
    const int n = w + 2 * j;
    const int g = n * 16 + gl;
    whh0F[j] = gload_bfrag(W_hh0, g, kb);
    whh1F[j] = gload_bfrag(W_hh1, g, kb);
    wih1F[j] = gload_bfrag(W_ih1, g, kb);
    wx[j]  = W_ih0[g];                   // I == 1
    bb0[j] = b_ih0[g] + b_hh0[g];
    const float b1 = b_ih1[g] + b_hh1[g];
    b1h[j] = (j == 2) ? b1 : 0.5f * b1;  // g-gate: plain tanh; others: sigmoid fold
  }
  const float wfc = W_fc[w * 16 + gl];

  // fragment lane offsets into the staging buffers (halfword units)
  const int aBase = gl * HID + ((lane < 16) ? 0 : 16);
  const int sCol  = w * 16 + gl;         // hidden column this wave produces

  // ---- recurrent state ----
  v16h hA0 = {}, hA1 = {};               // h in A-fragment layout (f16)
  float c0[8] = {}, c1[8] = {};          // cell state for this wave's hidden half
  float hv1[8] = {};

  for (int t = 0; t < TSTEPS; ++t) {
    __builtin_prefetch(xrow + t + 64, 0, 3);

    // broadcast x_t for the 8 rows this lane covers
    const float xv = xrow[t];
    float xr[8];
#pragma unroll
    for (int r = 0; r < 8; ++r) xr[r] = __shfl(xv, hb + r, 32);

    v8f acc[NGATE_TILES];

    // ============ layer 0: g = x*W_ih0 + b + h0 @ W_hh0^T ============
#pragma unroll
    for (int j = 0; j < NGATE_TILES; ++j) {
      v8f cinit;
#pragma unroll
      for (int r = 0; r < 8; ++r) cinit[r] = __builtin_fmaf(xr[r], wx[j], bb0[j]);
      acc[j] = wmma_f16(hA0, whh0F[j], cinit);
    }

#pragma unroll
    for (int r = 0; r < 8; ++r) {
      const float ig = sigmoid_fast(acc[0][r]);
      const float fg = sigmoid_fast(acc[1][r]);
      const float gg = tanh_fast(acc[2][r]);
      const float og = sigmoid_fast(acc[3][r]);
      const float c  = __builtin_fmaf(fg, c0[r], ig * gg);
      c0[r] = c;
      const float h = og * tanh_fast(c);
      sH0[(hb + r) * HID + sCol] = (f16)h;  // D-layout -> row-major
    }
    __syncthreads();                        // both waves' h0 halves visible
    hA0 = lds_load_afrag(&sH0[aBase]);      // transpose read back (full row)

    // ====== layer 1: g = h0_t @ W_ih1^T + h1_{t-1} @ W_hh1^T + b ======
#pragma unroll
    for (int j = 0; j < NGATE_TILES; ++j) {
      const v8f z = {};
      const v8f tmp = wmma_f16(hA1, whh1F[j], z);
      acc[j] = wmma_f16(hA0, wih1F[j], tmp);
    }

#pragma unroll
    for (int r = 0; r < 8; ++r) {
      // bias folded into the tanh argument (fma merges the +b)
      const float ig = __builtin_fmaf(
          0.5f, tanh_fast(__builtin_fmaf(0.5f, acc[0][r], b1h[0])), 0.5f);
      const float fg = __builtin_fmaf(
          0.5f, tanh_fast(__builtin_fmaf(0.5f, acc[1][r], b1h[1])), 0.5f);
      const float gg = tanh_fast(acc[2][r] + b1h[2]);
      const float og = __builtin_fmaf(
          0.5f, tanh_fast(__builtin_fmaf(0.5f, acc[3][r], b1h[3])), 0.5f);
      const float c  = __builtin_fmaf(fg, c1[r], ig * gg);
      c1[r] = c;
      const float h = og * tanh_fast(c);
      hv1[r] = h;
      sH1[(hb + r) * HID + sCol] = (f16)h;
    }
    __syncthreads();
    hA1 = lds_load_afrag(&sH1[aBase]);
  }

  // ============ final FC: out[b] = h1_last[b,:] . W_fc + b_fc ============
  float p[8];
#pragma unroll
  for (int r = 0; r < 8; ++r) p[r] = hv1[r] * wfc;
#pragma unroll
  for (int m = 1; m < 16; m <<= 1) {
#pragma unroll
    for (int r = 0; r < 8; ++r) p[r] += __shfl_xor(p[r], m, 32);
  }
  if (gl == 0) {
#pragma unroll
    for (int r = 0; r < 8; ++r) sRed[w][hb + r] = p[r];
  }
  __syncthreads();
  if (w == 0 && gl == 0) {
    const float bias = b_fc[0];
#pragma unroll
    for (int r = 0; r < 8; ++r)
      out[b0 + hb + r] = sRed[0][hb + r] + sRed[1][hb + r] + bias;
  }
}

extern "C" void kernel_launch(void* const* d_in, const int* in_sizes, int n_in,
                              void* d_out, int out_size, void* d_ws, size_t ws_size,
                              hipStream_t stream) {
  const float* x     = (const float*)d_in[0];
  const float* W_ih0 = (const float*)d_in[1];
  const float* W_hh0 = (const float*)d_in[2];
  const float* b_ih0 = (const float*)d_in[3];
  const float* b_hh0 = (const float*)d_in[4];
  const float* W_ih1 = (const float*)d_in[5];
  const float* W_hh1 = (const float*)d_in[6];
  const float* b_ih1 = (const float*)d_in[7];
  const float* b_hh1 = (const float*)d_in[8];
  const float* W_fc  = (const float*)d_in[9];
  const float* b_fc  = (const float*)d_in[10];

  dim3 grid(BATCH / TILE_B);   // 128 blocks, one batch tile each
  dim3 block(64);              // 2 cooperating waves (gate split)
  lstm2_fused_wmma<<<grid, block, 0, stream>>>(
      x, W_ih0, W_hh0, b_ih0, b_hh0, W_ih1, W_hh1, b_ih1, b_hh1, W_fc, b_fc,
      (float*)d_out);
}
